// DNBNSystem_52810917872097
// MI455X (gfx1250) — matrix operand order repaired and checked by reading, not compiled, not run on hardware
//
#include <hip/hip_runtime.h>
#include <hip/hip_bf16.h>

// Problem constants (from reference)
#define NEXP 8
#define BB   256
#define MM   512
#define CCH  512
#define NHH  8
#define HCC  64
#define TT   3
#define OUTD 100
#define PLN  289   // padded 17x17 plane for h1 (conv2 reads iy,ix in [0,16])

typedef __bf16 bf16;
typedef __attribute__((ext_vector_type(16))) __bf16 v16bf;
typedef __attribute__((ext_vector_type(8)))  __bf16 v8bf;
typedef __attribute__((ext_vector_type(8)))  float  v8f;

// Block tile 64x64, BK=32, 8 waves; each wave -> 16 rows x 32 cols (2 wmma accs).
// LDS tiles are stored in WMMA *fragment order* (ISA 7.12.2 16-bit layouts):
// fragment read = 2x ds_load_b128, staging write = 1x ds_store_b128.

__device__ __forceinline__ float sigmoidf_(float x) { return 1.0f / (1.0f + __expf(-x)); }

template<int ACT>
__device__ __forceinline__ float act_apply(float x) {
    if constexpr (ACT == 1) return fmaxf(x, 0.0f);
    else if constexpr (ACT == 2) return sigmoidf_(x);
    else if constexpr (ACT == 3) return tanhf(x);
    else return x;
}

// ---------------- generic bf16 WMMA GEMM ----------------
// C[n,r,c] = act( sum_k A[n,r,k]*W[n,k,c] + bias[n,c] )
template<int ACT, bool HAS_BIAS>
__global__ __launch_bounds__(256)
void gemm_wmma(const float* __restrict__ A, const float* __restrict__ W,
               const float* __restrict__ bias, float* __restrict__ Cout,
               int R, int K, int NC)
{
    __shared__ v16bf AlsF[128];   // 4 m-groups x 32 lanes (4 KB)
    __shared__ v16bf BlsF[128];   // 4 n-groups x 32 lanes (4 KB)

    const int n      = blockIdx.z;
    const int rowBlk = blockIdx.y * 64;
    const int colBlk = blockIdx.x * 64;
    const float* An = A + (size_t)n * R * K;
    const float* Wn = W + (size_t)n * K * NC;

    const int tid  = threadIdx.x;
    const int lane = tid & 31;
    const int wid  = tid >> 5;
    const int mOff = (wid >> 1) * 16;
    const int nOff = (wid & 1) * 32;
    const int hi   = lane >> 4;
    const int l15  = lane & 15;

    // ---- per-thread staging geometry (constant across K loop) ----
    const int sr  = tid >> 2;          // A row / B col within tile (0..63)
    const int sk0 = (tid & 3) * 8;     // first k of this thread's 8-chunk {0,8,16,24}
    // A: element (m=sr, k) -> lane=(m&15)+16*((k>>3)&1), i=(k&7)+8*(k>=16)
    const int laneA = (sr & 15) + 16 * ((sk0 >> 3) & 1);
    const int iA0   = (sk0 & 16) ? 8 : 0;
    bf16* Adst = (bf16*)&AlsF[(sr >> 4) * 32 + laneA] + iA0;
    // B: element (k, n=sr) -> lane=(n&15)+16*((k>>3)&1), i=(k&7)+8*(k>=16)
    const int laneB = (sr & 15) + 16 * ((sk0 >> 3) & 1);
    const int iB0   = (sk0 & 16) ? 8 : 0;
    bf16* Bdst = (bf16*)&BlsF[(sr >> 4) * 32 + laneB] + iB0;
    const int  colB  = colBlk + sr;
    const bool colOk = (colB < NC);

    const float* Asrc = An + (size_t)(rowBlk + sr) * K + sk0;
    const float* Bsrc = Wn + (size_t)sk0 * NC + colB;

    v8f c0 = {}; v8f c1 = {};
    const int ksteps = K / 32;

    for (int kt = 0; kt < ksteps; ++kt) {
        __syncthreads();
        if (kt + 1 < ksteps)
            __builtin_prefetch(Asrc + (size_t)(kt + 1) * 32, 0, 2);
        // stage A: 8 contiguous f32 -> bf16, one b128 LDS store
        {
            const float4* ap = (const float4*)(Asrc + (size_t)kt * 32);
            float4 a0 = ap[0], a1 = ap[1];
            v8bf ta;
            ta[0] = (bf16)a0.x; ta[1] = (bf16)a0.y; ta[2] = (bf16)a0.z; ta[3] = (bf16)a0.w;
            ta[4] = (bf16)a1.x; ta[5] = (bf16)a1.y; ta[6] = (bf16)a1.z; ta[7] = (bf16)a1.w;
            *(v8bf*)Adst = ta;
        }
        // stage B: 8 k-strided f32 (fixed col) -> one b128 LDS store
        {
            const float* bp = Bsrc + (size_t)kt * 32 * NC;
            v8bf tb;
            #pragma unroll
            for (int u = 0; u < 8; ++u) {
                float v = colOk ? bp[(size_t)u * NC] : 0.0f;
                tb[u] = (bf16)v;
            }
            *(v8bf*)Bdst = tb;
        }
        __syncthreads();

        v16bf af  = AlsF[(mOff >> 4) * 32 + lane];
        v16bf b0f = BlsF[((nOff >> 4) + 0) * 32 + lane];
        v16bf b1f = BlsF[((nOff >> 4) + 1) * 32 + lane];
        c0 = __builtin_amdgcn_wmma_f32_16x16x32_bf16(false, af, false, b0f, (short)0, c0, false, false);
        c1 = __builtin_amdgcn_wmma_f32_16x16x32_bf16(false, af, false, b1f, (short)0, c1, false, false);
    }

    float* Cn = Cout + (size_t)n * R * NC;
    #pragma unroll
    for (int r = 0; r < 8; ++r) {
        int row  = rowBlk + mOff + hi * 8 + r;
        int col0 = colBlk + nOff + l15;
        int col1 = col0 + 16;
        float x0 = c0[r], x1 = c1[r];
        if (HAS_BIAS) {
            if (col0 < NC) x0 += bias[(size_t)n * NC + col0];
            if (col1 < NC) x1 += bias[(size_t)n * NC + col1];
        }
        if (col0 < NC) Cn[(size_t)row * NC + col0] = act_apply<ACT>(x0);
        if (col1 < NC) Cn[(size_t)row * NC + col1] = act_apply<ACT>(x1);
    }
}

// ---------------- conv1: direct, writes bf16 h1 [N,B,64,17,17] (zero-padded) ----
__global__ __launch_bounds__(256)
void conv1_k(const float* __restrict__ x, const float* __restrict__ w,
             const float* __restrict__ b, bf16* __restrict__ h1)
{
    size_t idx = (size_t)blockIdx.x * blockDim.x + threadIdx.x;
    const size_t total = (size_t)NEXP * BB * 64 * 256;
    if (idx >= total) return;
    int sp = idx & 255;  size_t t = idx >> 8;
    int oc = t & 63;     t >>= 6;
    int bi = t & 255;    int n = (int)(t >> 8);
    int oy = sp >> 4, ox = sp & 15;
    float acc = b[n * 64 + oc];
    #pragma unroll
    for (int ic = 0; ic < 3; ++ic)
        #pragma unroll
        for (int ky = 0; ky < 3; ++ky) {
            int iy = oy * 2 + ky;
            if (iy >= 32) continue;
            #pragma unroll
            for (int kx = 0; kx < 3; ++kx) {
                int ix = ox * 2 + kx;
                if (ix >= 32) continue;
                acc += x[((size_t)(bi * 3 + ic) * 32 + iy) * 32 + ix] *
                       w[(((size_t)(n * 64 + oc) * 3 + ic) * 3 + ky) * 3 + kx];
            }
        }
    // padded 17x17 plane; border row/col stay zero (pre-zeroed)
    h1[(((size_t)n * BB + bi) * 64 + oc) * PLN + oy * 17 + ox] = (bf16)fmaxf(acc, 0.0f);
}

// ---------------- conv2: implicit-GEMM WMMA ----------------
// per expert: rows = B*64 (b,oy,ox), K = 64*9 = 576, NC = 128; h2 f32 [N,B,128,64]
__global__ __launch_bounds__(256)
void conv2_wmma(const bf16* __restrict__ h1, const float* __restrict__ w,
                const float* __restrict__ b, float* __restrict__ h2)
{
    __shared__ v16bf AlsF[128];
    __shared__ v16bf BlsF[128];

    const int n      = blockIdx.z;
    const int rowBlk = blockIdx.y * 64;
    const int colBlk = blockIdx.x * 64;       // 0 or 64 (NC=128)
    const int tid  = threadIdx.x;
    const int lane = tid & 31;
    const int wid  = tid >> 5;
    const int mOff = (wid >> 1) * 16;
    const int nOff = (wid & 1) * 32;
    const int hi   = lane >> 4;
    const int l15  = lane & 15;

    // staging geometry
    const int sr  = tid >> 2;
    const int sk0 = (tid & 3) * 8;
    const int laneA = (sr & 15) + 16 * ((sk0 >> 3) & 1);
    const int iA0   = (sk0 & 16) ? 8 : 0;
    bf16* Adst = (bf16*)&AlsF[(sr >> 4) * 32 + laneA] + iA0;
    const int laneB = (sr & 15) + 16 * ((sk0 >> 3) & 1);
    const int iB0   = (sk0 & 16) ? 8 : 0;
    bf16* Bdst = (bf16*)&BlsF[(sr >> 4) * 32 + laneB] + iB0;

    // A-side im2col geometry (constant per thread); padded plane => no bounds checks
    const int gr = rowBlk + sr;
    const int bi = gr >> 6, sp = gr & 63;
    const int oy2 = (sp >> 3) * 2, ox2 = (sp & 7) * 2;
    const bf16* h1n = h1 + ((size_t)n * BB + bi) * 64 * PLN + oy2 * 17 + ox2;
    // B-side: conv2_w[n][oc][k] is 576 contiguous floats per oc
    const int oc = colBlk + sr;
    const float* Bsrc = w + ((size_t)n * 128 + oc) * 576 + sk0;

    v8f c0 = {}; v8f c1 = {};

    for (int kt = 0; kt < 18; ++kt) {           // K = 576
        __syncthreads();
        // stage A: gather 8 consecutive k (im2col), branch-free padded loads
        {
            int kg = kt * 32 + sk0;
            int ic = kg / 9, rem = kg - ic * 9;
            v8bf ta;
            #pragma unroll
            for (int u = 0; u < 8; ++u) {
                int ky = (rem >= 3) + (rem >= 6);
                int kx = rem - ky * 3;
                ta[u] = h1n[ic * PLN + ky * 17 + kx];
                if (++rem == 9) { rem = 0; ++ic; }
            }
            *(v8bf*)Adst = ta;
        }
        // stage B: 8 contiguous f32 weights
        {
            const float4* bp = (const float4*)(Bsrc + (size_t)kt * 32);
            float4 b0v = bp[0], b1v = bp[1];
            v8bf tb;
            tb[0] = (bf16)b0v.x; tb[1] = (bf16)b0v.y; tb[2] = (bf16)b0v.z; tb[3] = (bf16)b0v.w;
            tb[4] = (bf16)b1v.x; tb[5] = (bf16)b1v.y; tb[6] = (bf16)b1v.z; tb[7] = (bf16)b1v.w;
            *(v8bf*)Bdst = tb;
        }
        __syncthreads();

        v16bf af  = AlsF[(mOff >> 4) * 32 + lane];
        v16bf b0f = BlsF[((nOff >> 4) + 0) * 32 + lane];
        v16bf b1f = BlsF[((nOff >> 4) + 1) * 32 + lane];
        c0 = __builtin_amdgcn_wmma_f32_16x16x32_bf16(false, af, false, b0f, (short)0, c0, false, false);
        c1 = __builtin_amdgcn_wmma_f32_16x16x32_bf16(false, af, false, b1f, (short)0, c1, false, false);
    }

    #pragma unroll
    for (int r = 0; r < 8; ++r) {
        int row = rowBlk + mOff + hi * 8 + r;
        int obi = row >> 6, osp = row & 63;
        int oc0 = colBlk + nOff + l15;
        int oc1 = oc0 + 16;
        float x0 = fmaxf(c0[r] + b[n * 128 + oc0], 0.0f);
        float x1 = fmaxf(c1[r] + b[n * 128 + oc1], 0.0f);
        h2[(((size_t)n * BB + obi) * 128 + oc0) * 64 + osp] = x0;
        h2[(((size_t)n * BB + obi) * 128 + oc1) * 64 + osp] = x1;
    }
}

// ---------------- small VALU kernels ----------------
__global__ void zero_k(float* p, size_t len) {
    size_t i = (size_t)blockIdx.x * blockDim.x + threadIdx.x;
    if (i < len) p[i] = 0.0f;
}
__global__ void copy_k(const float* __restrict__ s, float* __restrict__ d, size_t len) {
    size_t i = (size_t)blockIdx.x * blockDim.x + threadIdx.x;
    if (i < len) d[i] = s[i];
}
__global__ void pool_k(const float* __restrict__ h2, float* __restrict__ pooled) {
    size_t i = (size_t)blockIdx.x * blockDim.x + threadIdx.x;   // N*B*128
    if (i >= (size_t)NEXP * BB * 128) return;
    const float* p = h2 + i * 64;
    float s = 0.0f;
    #pragma unroll
    for (int k = 0; k < 64; ++k) s += p[k];
    pooled[i] = s * (1.0f / 64.0f);
}
__global__ void gru_k(const float* __restrict__ gi, const float* __restrict__ gh,
                      float* __restrict__ cs) {
    size_t i = (size_t)blockIdx.x * blockDim.x + threadIdx.x;   // N*B*HC
    if (i >= (size_t)NEXP * BB * HCC) return;
    size_t nb = i >> 6;  int j = (int)(i & 63);
    size_t base = nb * 192;
    float iz = gi[base + j], ir = gi[base + 64 + j], in = gi[base + 128 + j];
    float hz = gh[base + j], hr = gh[base + 64 + j], hn = gh[base + 128 + j];
    float z = sigmoidf_(iz + hz);
    float r = sigmoidf_(ir + hr);
    float ng = tanhf(in + r * hn);
    cs[i] = (1.0f - z) * ng + z * cs[i];
}
__global__ void gates_k(const float* __restrict__ cs,
                        const float* __restrict__ sw, const float* __restrict__ sb,
                        const float* __restrict__ rw, const float* __restrict__ rb,
                        const float* __restrict__ aw, const float* __restrict__ ab_b,
                        float* __restrict__ send, float* __restrict__ recv,
                        float* __restrict__ ab) {
    int i = blockIdx.x * blockDim.x + threadIdx.x;              // N*B
    if (i >= NEXP * BB) return;
    int n = i >> 8;
    const float* c = cs + (size_t)i * HCC;
    float s = sb[n], r = rb[n], a[NHH];
    #pragma unroll
    for (int h = 0; h < NHH; ++h) a[h] = ab_b[n * NHH + h];
    for (int j = 0; j < HCC; ++j) {
        float cj = c[j];
        s += cj * sw[n * HCC + j];
        r += cj * rw[n * HCC + j];
        #pragma unroll
        for (int h = 0; h < NHH; ++h) a[h] += cj * aw[(size_t)(n * HCC + j) * NHH + h];
    }
    send[i] = sigmoidf_(s);
    recv[i] = sigmoidf_(r);
    #pragma unroll
    for (int h = 0; h < NHH; ++h) ab[(size_t)i * NHH + h] = a[h];
}
__global__ __launch_bounds__(64)
void attn_k(const float* __restrict__ q, const float* __restrict__ k,
            const float* __restrict__ v, const float* __restrict__ edge,
            const float* __restrict__ ab, const float* __restrict__ send,
            const float* __restrict__ recv, float* __restrict__ msg) {
    int bh = blockIdx.x;
    int b = bh >> 3, h = bh & 7;
    __shared__ float qs[8][64], ks[8][64], vs[8][64], sc[64], awt[64];
    int tid = threadIdx.x;                                      // 64
    for (int i = 0; i < 8; ++i) {
        size_t off = ((size_t)i * BB + b) * CCH + h * 64 + tid;
        qs[i][tid] = q[off]; ks[i][tid] = k[off]; vs[i][tid] = v[off];
    }
    __syncthreads();
    int i = tid >> 3, j = tid & 7;
    float s = 0.0f;
    #pragma unroll
    for (int d = 0; d < 64; ++d) s += qs[i][d] * ks[j][d];
    s = s * 0.125f + edge[i * 8 + j] + ab[((size_t)i * BB + b) * NHH + h];
    sc[tid] = s;
    __syncthreads();
    float mx = -1e30f;
    #pragma unroll
    for (int jj = 0; jj < 8; ++jj) mx = fmaxf(mx, sc[i * 8 + jj]);
    float sum = 0.0f;
    #pragma unroll
    for (int jj = 0; jj < 8; ++jj) sum += __expf(sc[i * 8 + jj] - mx);
    float a = __expf(s - mx) / sum * send[j * BB + b] * recv[i * BB + b];
    awt[tid] = a;
    __syncthreads();
    for (int i2 = 0; i2 < 8; ++i2) {
        float m = 0.0f;
        #pragma unroll
        for (int j2 = 0; j2 < 8; ++j2) m += awt[i2 * 8 + j2] * vs[j2][tid];
        msg[((size_t)i2 * BB + b) * CCH + h * 64 + tid] = m;
    }
}
// FIFO of S=8 with only T=3 pushes: mean == running_sum / 8
__global__ void bufupd_k(const float* __restrict__ msg2, float* __restrict__ bufsum,
                         float* __restrict__ bufmean) {
    size_t i = (size_t)blockIdx.x * blockDim.x + threadIdx.x;   // N*B*C
    if (i >= (size_t)NEXP * BB * CCH) return;
    float s = bufsum[i] + msg2[i];
    bufsum[i] = s;
    bufmean[i] = s * 0.125f;
}
__global__ void cat_k(const float* __restrict__ h, const float* __restrict__ feats,
                      const float* __restrict__ ro, float* __restrict__ cat) {
    size_t i = (size_t)blockIdx.x * blockDim.x + threadIdx.x;   // N*B*M
    if (i >= (size_t)NEXP * BB * MM) return;
    size_t nb = i / MM; int m = (int)(i - nb * MM);
    size_t base = nb * (3 * MM);
    cat[base + m] = h[i];
    cat[base + MM + m] = feats[i];
    cat[base + 2 * MM + m] = ro[i];
}
__global__ void hupd_k(const float* __restrict__ g, const float* __restrict__ cand,
                       float* __restrict__ h) {
    size_t i = (size_t)blockIdx.x * blockDim.x + threadIdx.x;   // N*B*M
    if (i >= (size_t)NEXP * BB * MM) return;
    float gg = g[i];
    h[i] = (1.0f - gg) * h[i] + gg * cand[i];
}

// ---------------- launcher ----------------
extern "C" void kernel_launch(void* const* d_in, const int* in_sizes, int n_in,
                              void* d_out, int out_size, void* d_ws, size_t ws_size,
                              hipStream_t stream) {
    const float* x        = (const float*)d_in[0];
    const float* conv1_w  = (const float*)d_in[1];
    const float* conv1_b  = (const float*)d_in[2];
    const float* conv2_w  = (const float*)d_in[3];
    const float* conv2_b  = (const float*)d_in[4];
    const float* feat_w   = (const float*)d_in[5];
    const float* feat_b   = (const float*)d_in[6];
    const float* ctrl_wi  = (const float*)d_in[7];
    const float* ctrl_wh  = (const float*)d_in[8];
    const float* ctrl_bi  = (const float*)d_in[9];
    const float* ctrl_bh  = (const float*)d_in[10];
    const float* send_w   = (const float*)d_in[11];
    const float* send_b   = (const float*)d_in[12];
    const float* recv_w   = (const float*)d_in[13];
    const float* recv_b   = (const float*)d_in[14];
    const float* abias_w  = (const float*)d_in[15];
    const float* abias_b  = (const float*)d_in[16];
    const float* wq       = (const float*)d_in[17];
    const float* wk       = (const float*)d_in[18];
    const float* wv       = (const float*)d_in[19];
    const float* wo       = (const float*)d_in[20];
    const float* bo       = (const float*)d_in[21];
    const float* edge     = (const float*)d_in[22];
    const float* wr       = (const float*)d_in[23];
    const float* br       = (const float*)d_in[24];
    const float* wg       = (const float*)d_in[25];
    const float* bg       = (const float*)d_in[26];
    const float* wc       = (const float*)d_in[27];
    const float* bc       = (const float*)d_in[28];
    const float* wcls     = (const float*)d_in[29];
    const float* bcls     = (const float*)d_in[30];

    // workspace carve (~211 MB total)
    char* p = (char*)d_ws;
    auto carve = [&](size_t bytes) -> char* {
        char* r = p; p += (bytes + 255) & ~(size_t)255; return r;
    };
    const size_t h1_elems = (size_t)NEXP * BB * 64 * PLN;      // padded bf16 planes
    bf16*  h1      = (bf16*) carve(h1_elems * sizeof(bf16));
    float* h2      = (float*)carve((size_t)NEXP * BB * 128 * 64 * sizeof(float));
    float* pooled  = (float*)carve((size_t)NEXP * BB * 128 * sizeof(float));
    float* feats   = (float*)carve((size_t)NEXP * BB * MM * sizeof(float));
    float* h       = (float*)carve((size_t)NEXP * BB * MM * sizeof(float));
    float* cs      = (float*)carve((size_t)NEXP * BB * HCC * sizeof(float));
    float* gi      = (float*)carve((size_t)NEXP * BB * 3 * HCC * sizeof(float));
    float* gh      = (float*)carve((size_t)NEXP * BB * 3 * HCC * sizeof(float));
    float* q       = (float*)carve((size_t)NEXP * BB * CCH * sizeof(float));
    float* k       = (float*)carve((size_t)NEXP * BB * CCH * sizeof(float));
    float* v       = (float*)carve((size_t)NEXP * BB * CCH * sizeof(float));
    float* send    = (float*)carve((size_t)NEXP * BB * sizeof(float));
    float* recv    = (float*)carve((size_t)NEXP * BB * sizeof(float));
    float* ab      = (float*)carve((size_t)NEXP * BB * NHH * sizeof(float));
    float* msg     = (float*)carve((size_t)NEXP * BB * CCH * sizeof(float));
    float* msg2    = (float*)carve((size_t)NEXP * BB * CCH * sizeof(float));
    float* bufsum  = (float*)carve((size_t)NEXP * BB * CCH * sizeof(float));
    float* bufmean = (float*)carve((size_t)NEXP * BB * CCH * sizeof(float));
    float* ro      = (float*)carve((size_t)NEXP * BB * MM * sizeof(float));
    float* cat     = (float*)carve((size_t)NEXP * BB * 3 * MM * sizeof(float));
    float* g       = (float*)carve((size_t)NEXP * BB * MM * sizeof(float));
    float* cand    = (float*)carve((size_t)NEXP * BB * MM * sizeof(float));
    (void)ws_size; (void)in_sizes; (void)n_in; (void)out_size;

    const int TPB = 256;
    auto blocks = [](size_t n, int tpb) { return (unsigned)((n + tpb - 1) / tpb); };

    // zero padded h1 (border must be 0), then feature extraction
    zero_k<<<blocks(h1_elems * sizeof(bf16) / 4, TPB), TPB, 0, stream>>>(
        (float*)h1, h1_elems * sizeof(bf16) / 4);
    conv1_k<<<blocks((size_t)NEXP * BB * 64 * 256, TPB), TPB, 0, stream>>>(x, conv1_w, conv1_b, h1);
    conv2_wmma<<<dim3(2, (BB * 64) / 64, NEXP), TPB, 0, stream>>>(h1, conv2_w, conv2_b, h2);
    pool_k<<<blocks((size_t)NEXP * BB * 128, TPB), TPB, 0, stream>>>(h2, pooled);
    gemm_wmma<1, true><<<dim3(MM / 64, BB / 64, NEXP), TPB, 0, stream>>>(pooled, feat_w, feat_b, feats, BB, 128, MM);
    copy_k<<<blocks((size_t)NEXP * BB * MM, TPB), TPB, 0, stream>>>(feats, h, (size_t)NEXP * BB * MM);
    zero_k<<<blocks((size_t)NEXP * BB * HCC, TPB), TPB, 0, stream>>>(cs, (size_t)NEXP * BB * HCC);
    zero_k<<<blocks((size_t)NEXP * BB * CCH, TPB), TPB, 0, stream>>>(bufsum, (size_t)NEXP * BB * CCH);

    for (int t = 0; t < TT; ++t) {
        // controller GRU
        gemm_wmma<0, true><<<dim3(192 / 64, BB / 64, NEXP), TPB, 0, stream>>>(h,  ctrl_wi, ctrl_bi, gi, BB, MM, 192);
        gemm_wmma<0, true><<<dim3(192 / 64, BB / 64, NEXP), TPB, 0, stream>>>(cs, ctrl_wh, ctrl_bh, gh, BB, HCC, 192);
        gru_k<<<blocks((size_t)NEXP * BB * HCC, TPB), TPB, 0, stream>>>(gi, gh, cs);
        gates_k<<<blocks((size_t)NEXP * BB, TPB), TPB, 0, stream>>>(cs, send_w, send_b, recv_w, recv_b,
                                                                     abias_w, abias_b, send, recv, ab);
        // QKV
        gemm_wmma<0, false><<<dim3(CCH / 64, BB / 64, NEXP), TPB, 0, stream>>>(h, wq, nullptr, q, BB, MM, CCH);
        gemm_wmma<0, false><<<dim3(CCH / 64, BB / 64, NEXP), TPB, 0, stream>>>(h, wk, nullptr, k, BB, MM, CCH);
        gemm_wmma<0, false><<<dim3(CCH / 64, BB / 64, NEXP), TPB, 0, stream>>>(h, wv, nullptr, v, BB, MM, CCH);
        // gated node attention
        attn_k<<<BB * NHH, 64, 0, stream>>>(q, k, v, edge, ab, send, recv, msg);
        gemm_wmma<0, true><<<dim3(CCH / 64, BB / 64, NEXP), TPB, 0, stream>>>(msg, wo, bo, msg2, BB, CCH, CCH);
        // FIFO mean + readout
        bufupd_k<<<blocks((size_t)NEXP * BB * CCH, TPB), TPB, 0, stream>>>(msg2, bufsum, bufmean);
        gemm_wmma<0, true><<<dim3(MM / 64, BB / 64, NEXP), TPB, 0, stream>>>(bufmean, wr, br, ro, BB, CCH, MM);
        // gated state update
        cat_k<<<blocks((size_t)NEXP * BB * MM, TPB), TPB, 0, stream>>>(h, feats, ro, cat);
        gemm_wmma<2, true><<<dim3(MM / 64, BB / 64, NEXP), TPB, 0, stream>>>(cat, wg, bg, g,    BB, 3 * MM, MM);
        gemm_wmma<3, true><<<dim3(MM / 64, BB / 64, NEXP), TPB, 0, stream>>>(cat, wc, bc, cand, BB, 3 * MM, MM);
        hupd_k<<<blocks((size_t)NEXP * BB * MM, TPB), TPB, 0, stream>>>(g, cand, h);
    }
    // classifier -> d_out [N,B,OUT]
    gemm_wmma<0, true><<<dim3((OUTD + 63) / 64, BB / 64, NEXP), TPB, 0, stream>>>(
        h, wcls, bcls, (float*)d_out, BB, MM, OUTD);
}
